// BaseBiLinear_66408784331155
// MI455X (gfx1250) — compile-verified, matrix operand order checked by reading, not run on hardware
//
#include <hip/hip_runtime.h>
#include <math.h>

typedef __bf16 bf16;
typedef __attribute__((ext_vector_type(16))) __bf16 v16bf;
typedef __attribute__((ext_vector_type(8)))  __bf16 v8bf;
typedef __attribute__((ext_vector_type(8)))  float  v8f;
typedef int v4i __attribute__((vector_size(16)));

#define BATCH 2
#define NCAM  6
#define CIN   256
#define MID   80
#define FH    16
#define FW    44
#define FWH   50      // x-halo: col 0 zero, data at 1..44, zeros 45..49
#define VZ    16
#define VY    128
#define VX    128
#define VXH   130     // x-halo: col 0 zero, data at 1..128, col 129 zero
#define CPAD  96

// ---- async-to-LDS availability probe ----
#if defined(__has_builtin)
#if __has_builtin(__builtin_amdgcn_global_load_async_to_lds_b128)
#define HAVE_ASYNC_LDS 1
#endif
#endif
#ifndef HAVE_ASYNC_LDS
#define HAVE_ASYNC_LDS 0
#endif

// ---- workspace layout (bytes) ----
#define OFF_PMAT   0ull
#define OFF_WLOW   0x100000ull                 // 5*9*8*32*16 bf16 = 368640 B
#define OFF_WBASE  0x180000ull                 // 27*5*3*32*16 bf16 = 414720 B
#define OFF_WFEAT  0x200000ull
#define OFF_WSRD   0x280000ull                 // 27*2*3*32*16 bf16
#define OFF_WOUT   0x300000ull                 // 5*40*32*16 bf16
#define OFF_IMGCL  0x400000ull                 // 12*16*50*256 bf16 = 4.9 MB
#define OFF_XCL    0x900000ull                 // 12*16*44*80 f32 = 2.7 MB
#define OFF_VOX    0xC00000ull                 // 2*16*128*130*96 bf16 = 102236160 B
#define VOXBYTES   102236160ull
#define OFF_BASECL (OFF_VOX + VOXBYTES)
#define OFF_SRD    (OFF_BASECL + VOXBYTES)     // 2*16*128*128*32 bf16 = 32 MB
#define OFF_FEAT   OFF_VOX                     // reuse: vox dead after base conv

// ---------------- bf16 helpers ----------------
union bfu { unsigned short u; bf16 b; };
__device__ __forceinline__ bf16 f2bf(float f) {
  unsigned x = __float_as_uint(f);
  x += 0x7fffu + ((x >> 16) & 1u);
  bfu t; t.u = (unsigned short)(x >> 16); return t.b;
}
__device__ __forceinline__ float bf2f(bf16 b) {
  bfu t; t.b = b; return __uint_as_float(((unsigned)t.u) << 16);
}
__device__ __forceinline__ bf16 bf_zero() { bfu t; t.u = 0; return t.b; }

// ---------------- WMMA fragment helpers ----------------
// 16-bit A/B fragment per lane: row/col = lane&15, kb = (lane>>4)*8;
// elements 0..7 hold K=kb..kb+7, elements 8..15 hold K=16+kb..16+kb+7
__device__ __forceinline__ v16bf load_frag_row(const bf16* chan_base, int kb) {
  v8bf lo = *(const v8bf*)(chan_base + kb);
  v8bf hi = *(const v8bf*)(chan_base + 16 + kb);
  v16bf r;
#pragma unroll
  for (int i = 0; i < 8; ++i) { r[i] = lo[i]; r[i + 8] = hi[i]; }
  return r;
}
__device__ __forceinline__ v8f wmma_bf16(v16bf a, v16bf b, v8f c) {
  return __builtin_amdgcn_wmma_f32_16x16x32_bf16(false, a, false, b, (short)0, c,
                                                 false, false);
}

// cooperative global -> LDS stage (async path if toolchain exposes it)
__device__ __forceinline__ void stage_to_lds(const bf16* __restrict__ g,
                                             bf16* __restrict__ s, int bytes,
                                             int tid, int nth) {
  const int nv = bytes >> 4;
#if HAVE_ASYNC_LDS
  for (int i = tid; i < nv; i += nth) {
    __builtin_amdgcn_global_load_async_to_lds_b128(
        (v4i*)((char*)const_cast<bf16*>(g) + i * 16),
        (__attribute__((address_space(3))) v4i*)((char*)s + i * 16), 0, 0);
  }
#if defined(__has_builtin) && __has_builtin(__builtin_amdgcn_s_wait_asynccnt)
  __builtin_amdgcn_s_wait_asynccnt(0);
#else
  asm volatile("s_wait_asynccnt 0x0" ::: "memory");
#endif
#else
  const uint4* gv = (const uint4*)g;
  uint4* sv = (uint4*)s;
  for (int i = tid; i < nv; i += nth) sv[i] = gv[i];
#endif
}

// ---------------- small math ----------------
__device__ __forceinline__ float sigm(float x) { return 1.f / (1.f + expf(-x)); }
__device__ __forceinline__ float softplus100(float v) {
  return fmaxf(v, 0.f) + log1pf(expf(-fabsf(v * 100.f))) * 0.01f;
}

__device__ void mat4mul(const float* A, const float* B, float* C) {
#pragma unroll
  for (int i = 0; i < 4; ++i)
#pragma unroll
    for (int j = 0; j < 4; ++j) {
      float s = 0.f;
#pragma unroll
      for (int k = 0; k < 4; ++k) s += A[i * 4 + k] * B[k * 4 + j];
      C[i * 4 + j] = s;
    }
}
__device__ void inv4(const float* m, float* o) {
  float inv[16];
  inv[0]= m[5]*m[10]*m[15]-m[5]*m[11]*m[14]-m[9]*m[6]*m[15]+m[9]*m[7]*m[14]+m[13]*m[6]*m[11]-m[13]*m[7]*m[10];
  inv[4]=-m[4]*m[10]*m[15]+m[4]*m[11]*m[14]+m[8]*m[6]*m[15]-m[8]*m[7]*m[14]-m[12]*m[6]*m[11]+m[12]*m[7]*m[10];
  inv[8]= m[4]*m[9]*m[15]-m[4]*m[11]*m[13]-m[8]*m[5]*m[15]+m[8]*m[7]*m[13]+m[12]*m[5]*m[11]-m[12]*m[7]*m[9];
  inv[12]=-m[4]*m[9]*m[14]+m[4]*m[10]*m[13]+m[8]*m[5]*m[14]-m[8]*m[6]*m[13]-m[12]*m[5]*m[10]+m[12]*m[6]*m[9];
  inv[1]=-m[1]*m[10]*m[15]+m[1]*m[11]*m[14]+m[9]*m[2]*m[15]-m[9]*m[3]*m[14]-m[13]*m[2]*m[11]+m[13]*m[3]*m[10];
  inv[5]= m[0]*m[10]*m[15]-m[0]*m[11]*m[14]-m[8]*m[2]*m[15]+m[8]*m[3]*m[14]+m[12]*m[2]*m[11]-m[12]*m[3]*m[10];
  inv[9]=-m[0]*m[9]*m[15]+m[0]*m[11]*m[13]+m[8]*m[1]*m[15]-m[8]*m[3]*m[13]-m[12]*m[1]*m[11]+m[12]*m[3]*m[9];
  inv[13]= m[0]*m[9]*m[14]-m[0]*m[10]*m[13]-m[8]*m[1]*m[14]+m[8]*m[2]*m[13]+m[12]*m[1]*m[10]-m[12]*m[2]*m[9];
  inv[2]= m[1]*m[6]*m[15]-m[1]*m[7]*m[14]-m[5]*m[2]*m[15]+m[5]*m[3]*m[14]+m[13]*m[2]*m[7]-m[13]*m[3]*m[6];
  inv[6]=-m[0]*m[6]*m[15]+m[0]*m[7]*m[14]+m[4]*m[2]*m[15]-m[4]*m[3]*m[14]-m[12]*m[2]*m[7]+m[12]*m[3]*m[6];
  inv[10]= m[0]*m[5]*m[15]-m[0]*m[7]*m[13]-m[4]*m[1]*m[15]+m[4]*m[3]*m[13]+m[12]*m[1]*m[7]-m[12]*m[3]*m[5];
  inv[14]=-m[0]*m[5]*m[14]+m[0]*m[6]*m[13]+m[4]*m[1]*m[14]-m[4]*m[2]*m[13]-m[12]*m[1]*m[6]+m[12]*m[2]*m[5];
  inv[3]=-m[1]*m[6]*m[11]+m[1]*m[7]*m[10]+m[5]*m[2]*m[11]-m[5]*m[3]*m[10]-m[9]*m[2]*m[7]+m[9]*m[3]*m[6];
  inv[7]= m[0]*m[6]*m[11]-m[0]*m[7]*m[10]-m[4]*m[2]*m[11]+m[4]*m[3]*m[10]+m[8]*m[2]*m[7]-m[8]*m[3]*m[6];
  inv[11]=-m[0]*m[5]*m[11]+m[0]*m[7]*m[9]+m[4]*m[1]*m[11]-m[4]*m[3]*m[9]-m[8]*m[1]*m[7]+m[8]*m[3]*m[5];
  inv[15]= m[0]*m[5]*m[10]-m[0]*m[6]*m[9]-m[4]*m[1]*m[10]+m[4]*m[2]*m[9]+m[8]*m[1]*m[6]-m[8]*m[2]*m[5];
  float det = m[0]*inv[0]+m[1]*inv[4]+m[2]*inv[8]+m[3]*inv[12];
  float r = 1.f / det;
#pragma unroll
  for (int i = 0; i < 16; ++i) o[i] = inv[i] * r;
}

// ---------------- projection matrices ----------------
__global__ void k_build_proj(const float* s2e, const float* intr, const float* bda,
                             float* pmat) {
  int t = threadIdx.x;
  if (t >= BATCH * NCAM) return;
  int b = t / NCAM, n = t % NCAM;
  float invS[16], invB[16], tmp[16], P[16];
  inv4(s2e + (size_t)(b * NCAM + n) * 16, invS);
  inv4(bda + (size_t)b * 16, invB);
  mat4mul(invS, invB, tmp);
  mat4mul(intr + (size_t)(b * NCAM + n) * 16, tmp, P);
#pragma unroll
  for (int i = 0; i < 16; ++i) pmat[t * 16 + i] = P[i];
}

// ---------------- img_feats -> channels-last bf16 (with x halo) -------------
__global__ void k_img_cl(const float* img, bf16* dst) {
  int tid = blockIdx.x * blockDim.x + threadIdx.x;
  const int total = BATCH * NCAM * FH * FW * CIN;
  if (tid >= total) return;
  int c = tid % CIN; int r = tid / CIN;
  int x = r % FW; r /= FW;
  int y = r % FH; int im = r / FH;
  dst[(((size_t)im * FH + y) * FWH + (x + 1)) * CIN + c] =
      f2bf(img[(((size_t)im * CIN + c) * FH + y) * FW + x]);
}
__global__ void k_img_halo(bf16* buf) {
  int tid = blockIdx.x * blockDim.x + threadIdx.x;
  const int total = BATCH * NCAM * FH * 6 * CIN; // cols {0,45..49}
  if (tid >= total) return;
  int c = tid % CIN; int r = tid / CIN;
  int xi = r % 6; r /= 6;
  int y = r % FH; int im = r / FH;
  int xcol = (xi == 0) ? 0 : (44 + xi);
  buf[(((size_t)im * FH + y) * FWH + xcol) * CIN + c] = bf_zero();
}
// zero x-halo columns of a [planes][w][cpad] channels-last volume
__global__ void k_halo_zero(bf16* buf, int planes, int w, int cpad) {
  int tid = blockIdx.x * blockDim.x + threadIdx.x;
  int total = planes * 2 * cpad;
  if (tid >= total) return;
  int c = tid % cpad; int r = tid / cpad;
  int side = r & 1; int p = r >> 1;
  buf[((size_t)p * w + (side ? (w - 1) : 0)) * cpad + c] = bf_zero();
}

// ---------------- weight packing (fragment layout) ----------------
__device__ __forceinline__ int frag_K(int lane, int e) {
  int kb = (lane >> 4) * 8;
  return (e < 8) ? (kb + e) : (16 + kb + (e - 8));
}
// conv3d weights -> [tap][cot][kc][lane][16]  (tap-major for contiguous staging)
__global__ void k_pack_c3d(const float* w, bf16* dst) {
  int tid = blockIdx.x * blockDim.x + threadIdx.x;
  const int total = 5 * 27 * 3 * 32 * 16;
  if (tid >= total) return;
  int e = tid & 15; int lane = (tid >> 4) & 31; int r = tid >> 9;
  int kc = r % 3; r /= 3; int tap = r % 27; int cot = r / 27;
  int co = cot * 16 + (lane & 15);
  int ci = kc * 32 + frag_K(lane, e);
  float v = (ci < MID) ? w[((size_t)(co * MID + ci)) * 27 + tap] : 0.f;
  dst[((((size_t)tap * 5 + cot) * 3 + kc) * 32 + lane) * 16 + e] = f2bf(v);
}
__global__ void k_pack_srd(const float* wseg, const float* wden, const float* wrgb,
                           bf16* dst) {
  int tid = blockIdx.x * blockDim.x + threadIdx.x;
  const int total = 2 * 27 * 3 * 32 * 16;
  if (tid >= total) return;
  int e = tid & 15; int lane = (tid >> 4) & 31; int r = tid >> 9;
  int kc = r % 3; r /= 3; int tap = r % 27; int cot = r / 27;
  int row = cot * 16 + (lane & 15);
  int ci = kc * 32 + frag_K(lane, e);
  float v = 0.f;
  if (ci < MID) {
    if (row < 17)       v = wseg[((size_t)(row * MID + ci)) * 27 + tap];
    else if (row == 17) v = wden[(size_t)ci * 27 + tap];
    else if (row < 21)  v = wrgb[((size_t)((row - 18) * MID + ci)) * 27 + tap];
  }
  dst[((((size_t)tap * 2 + cot) * 3 + kc) * 32 + lane) * 16 + e] = f2bf(v);
}
__global__ void k_pack_lower(const float* w, bf16* dst) { // [cot][tap][kc][lane][16]
  int tid = blockIdx.x * blockDim.x + threadIdx.x;
  const int total = 5 * 9 * 8 * 32 * 16;
  if (tid >= total) return;
  int e = tid & 15; int lane = (tid >> 4) & 31; int r = tid >> 9;
  int kc = r % 8; r /= 8; int tap = r % 9; int cot = r / 9;
  int co = cot * 16 + (lane & 15);
  int ci = kc * 32 + frag_K(lane, e);
  dst[tid] = f2bf(w[((size_t)(co * CIN + ci)) * 9 + tap]);
}
__global__ void k_pack_out(const float* w, bf16* dst) { // [cot][kc][lane][16]
  int tid = blockIdx.x * blockDim.x + threadIdx.x;
  const int total = 5 * 40 * 32 * 16;
  if (tid >= total) return;
  int e = tid & 15; int lane = (tid >> 4) & 31; int r = tid >> 9;
  int kc = r % 40; int cot = r / 40;
  int co = cot * 16 + (lane & 15);
  int c = kc * 32 + frag_K(lane, e);
  dst[tid] = f2bf(w[(size_t)co * (MID * VZ) + c]);
}

// ---------------- channel_lower conv2d (WMMA, branchless inner loop) --------
__global__ __launch_bounds__(256) void k_conv_lower(const bf16* imgcl,
                                                    const bf16* wpk, float* xcl) {
  int wv = (blockIdx.x * blockDim.x + threadIdx.x) >> 5;
  int lane = threadIdx.x & 31;
  const int NW = BATCH * NCAM * 5 * FH * 3; // 2880 waves
  if (wv >= NW) return;
  int xt = wv % 3; int r = wv / 3;
  int y = r % FH; r /= FH;
  int cot = r % 5; int im = r / 5;
  int col = lane & 15, kb = (lane >> 4) * 8, rowb = kb;

  v8f acc;
#pragma unroll
  for (int i = 0; i < 8; ++i) acc[i] = 0.f;

  for (int kh = 0; kh < 3; ++kh) {
    int yy = y + kh - 1;
    if (yy < 0 || yy >= FH) continue; // wave-uniform; zero-pad contribution
    for (int kw = 0; kw < 3; ++kw) {
      const bf16* row =
          imgcl + ((size_t)(im * FH + yy) * FWH + (xt * 16 + col + kw)) * CIN;
      int tap = kh * 3 + kw;
#pragma unroll
      for (int kc = 0; kc < 8; ++kc) {
        v16bf bf = load_frag_row(row + kc * 32, kb);
        const v16bf* ap = (const v16bf*)(wpk +
            ((((size_t)cot * 9 + tap) * 8 + kc) * 32 + lane) * 16);
        acc = wmma_bf16(*ap, bf, acc);
      }
    }
  }
  int xx = xt * 16 + col;
  if (xx < FW) {
#pragma unroll
    for (int v = 0; v < 8; ++v) {
      int co = cot * 16 + rowb + v;
      xcl[((size_t)(im * FH + y) * FW + xx) * MID + co] = acc[v];
    }
  }
}

// ---------------- project + bilinear gather -> vox ----------------
__device__ __forceinline__ void corner_acc(const float* xcl, int im, int vi, int ui,
                                           int c0, float w, float* acc) {
  if (vi < 0 || vi >= FH || ui < 0 || ui >= FW) return;
  const float* p = xcl + ((size_t)(im * FH + vi) * FW + ui) * MID + c0;
#pragma unroll
  for (int j = 0; j < 16; ++j) acc[j] += w * p[j];
}

__global__ void k_gather(const float* xcl, const float* pmat, const float* ida,
                         bf16* vox) {
  int tid = blockIdx.x * blockDim.x + threadIdx.x;
  const int total = BATCH * VZ * VY * VX * 6;
  if (tid >= total) return;
  int chunk = tid % 6; int id = tid / 6;
  int vx = id & 127; id >>= 7;
  int vy = id & 127; id >>= 7;
  int vz = id & 15;  id >>= 4;
  int b = id;
  bf16* dst = vox +
      ((size_t)((b * VZ + vz) * VY + vy) * VXH + (vx + 1)) * CPAD + chunk * 16;
  if (chunk == 5) { // zero pad channels 80..95
#pragma unroll
    for (int j = 0; j < 16; ++j) dst[j] = bf_zero();
    return;
  }
  float px = -50.8f + 0.8f * vx;
  float py = -50.8f + 0.8f * vy;
  float pz = -4.75f + 0.5f * vz;
  float acc[16];
#pragma unroll
  for (int j = 0; j < 16; ++j) acc[j] = 0.f;
  float cnt = 0.f;
  int c0 = chunk * 16;
  for (int n = 0; n < NCAM; ++n) {
    const float* P = pmat + (size_t)(b * NCAM + n) * 16;
    float X = P[0] * px + P[1] * py + P[2] * pz + P[3];
    float Y = P[4] * px + P[5] * py + P[6] * pz + P[7];
    float Z = P[8] * px + P[9] * py + P[10] * pz + P[11];
    float zs = (fabsf(Z) > 1e-3f) ? Z : 1e-3f;
    float u = X / zs, v = Y / zs;
    const float* I = ida + (size_t)(b * NCAM + n) * 16;
    float uu = I[0] * u + I[1] * v + I[2] + I[3];
    float vv = I[4] * u + I[5] * v + I[6] + I[7];
    float uf = uu * (1.f / 16.f), vf = vv * (1.f / 16.f);
    bool valid = (Z > 1e-3f) && uf >= 0.f && uf <= (float)(FW - 1) && vf >= 0.f &&
                 vf <= (float)(FH - 1);
    if (!valid) continue;
    cnt += 1.f;
    float u0 = floorf(uf), v0 = floorf(vf);
    float du = uf - u0, dv = vf - v0;
    int u0i = (int)u0, v0i = (int)v0;
    int im = b * NCAM + n;
    corner_acc(xcl, im, v0i,     u0i,     c0, (1.f - dv) * (1.f - du), acc);
    corner_acc(xcl, im, v0i,     u0i + 1, c0, (1.f - dv) * du,         acc);
    corner_acc(xcl, im, v0i + 1, u0i,     c0, dv * (1.f - du),         acc);
    corner_acc(xcl, im, v0i + 1, u0i + 1, c0, dv * du,                 acc);
  }
  float s = 1.f / fmaxf(cnt, 1.f);
#pragma unroll
  for (int j = 0; j < 16; ++j) dst[j] = f2bf(acc[j] * s);
}

// ---------------- 3x3x3 conv3d via WMMA + LDS-staged weights ----------------
// All 8 waves of a block share (b,z,y) and differ in x-tile -> weights are
// cooperatively staged in LDS per (kd,kh) slice (3 kw taps at once).
// MODE 0: base (+b_base, softplus, CPAD halo layout, zero pad chans)
// MODE 1: srd  (seg/density/rgb bias, 32-ch layout)
// MODE 2: feat (+b_feat, [b][y][x][mid*16+z] layout)
template <int NCOT, int MODE>
__global__ __launch_bounds__(256) void k_conv3d(
    const bf16* __restrict__ in, const bf16* __restrict__ wpk, const float* bias0,
    const float* bias1, const float* bias2, bf16* __restrict__ out) {
  __shared__ alignas(32) bf16 smw[3 * NCOT * 3 * 32 * 16]; // 3 kw taps staged
  int tid = threadIdx.x;
  int lane = tid & 31;
  int wv = blockIdx.x * 8 + (tid >> 5);
  const int NW = BATCH * VZ * VY * (VX / 16);
  if (wv >= NW) return;
  int xt = wv & 7; int r = wv >> 3;
  int y = r & 127; r >>= 7;
  int z = r & 15; int b = r >> 4;
  int col = lane & 15, kb = (lane >> 4) * 8, rowb = kb;

  v8f acc[NCOT];
#pragma unroll
  for (int c = 0; c < NCOT; ++c)
#pragma unroll
    for (int i = 0; i < 8; ++i) acc[c][i] = 0.f;

  const int TAPELEMS = NCOT * 3 * 32 * 16; // one tap's packed frags
  for (int kd = 0; kd < 3; ++kd) {
    int zz = z + kd - 1;
    if (zz < 0 || zz >= VZ) continue; // uniform across block
    for (int kh = 0; kh < 3; ++kh) {
      int yy = y + kh - 1;
      if (yy < 0 || yy >= VY) continue; // uniform across block
      int tap0 = (kd * 3 + kh) * 3;
      __syncthreads(); // previous slice fully consumed
      stage_to_lds(wpk + (size_t)tap0 * TAPELEMS, smw, 3 * TAPELEMS * 2, tid, 256);
      __syncthreads(); // slice visible to all waves
      const bf16* rowb0 =
          in + ((size_t)((b * VZ + zz) * VY + yy) * VXH + (xt * 16 + col)) * CPAD;
      __builtin_prefetch(rowb0 + VXH * CPAD, 0, 0);
#pragma unroll
      for (int kw = 0; kw < 3; ++kw) {
        const bf16* row = rowb0 + (size_t)kw * CPAD;
#pragma unroll
        for (int kc = 0; kc < 3; ++kc) {
          v16bf bf = load_frag_row(row + kc * 32, kb);
#pragma unroll
          for (int c = 0; c < NCOT; ++c) {
            const v16bf* ap = (const v16bf*)(smw +
                (((size_t)(kw * NCOT + c) * 3 + kc) * 32 + lane) * 16);
            acc[c] = wmma_bf16(*ap, bf, acc[c]);
          }
        }
      }
    }
  }

  int xx = xt * 16 + col;
  if (MODE == 0) {
    size_t vo = ((size_t)((b * VZ + z) * VY + y) * VXH + xx + 1) * CPAD;
#pragma unroll
    for (int c = 0; c < NCOT; ++c)
#pragma unroll
      for (int v = 0; v < 8; ++v) {
        int co = c * 16 + rowb + v;
        out[vo + co] = f2bf(softplus100(acc[c][v] + bias0[co]));
      }
#pragma unroll
    for (int v = 0; v < 8; ++v) out[vo + 80 + rowb + v] = bf_zero();
  } else if (MODE == 1) {
    size_t vo = ((size_t)((b * VZ + z) * VY + y) * VX + xx) * 32;
#pragma unroll
    for (int c = 0; c < NCOT; ++c)
#pragma unroll
      for (int v = 0; v < 8; ++v) {
        int rr = c * 16 + rowb + v;
        float bv = 0.f;
        if (rr < 17) bv = bias0[rr];
        else if (rr == 17) bv = bias1[0];
        else if (rr < 21) bv = bias2[rr - 18];
        out[vo + rr] = f2bf(acc[c][v] + bv);
      }
  } else { // MODE 2
    size_t pb = ((size_t)(b * VY + y) * VX + xx) * (MID * VZ);
#pragma unroll
    for (int c = 0; c < NCOT; ++c)
#pragma unroll
      for (int v = 0; v < 8; ++v) {
        int co = c * 16 + rowb + v;
        out[pb + co * VZ + z] = f2bf(acc[c][v] + bias0[co]);
      }
  }
}

// ---------------- volumetric compositing ----------------
__global__ void k_composite(const bf16* srd, float* out) {
  int tid = blockIdx.x * blockDim.x + threadIdx.x;
  const int total = BATCH * VY * VX;
  if (tid >= total) return;
  int x = tid & 127, y = (tid >> 7) & 127, b = tid >> 14;
  float seg[17], rgb[3], trans = 1.f;
#pragma unroll
  for (int c = 0; c < 17; ++c) seg[c] = 0.f;
  rgb[0] = rgb[1] = rgb[2] = 0.f;
  for (int z = 0; z < VZ; ++z) {
    const bf16* row = srd + ((size_t)((b * VZ + z) * VY + y) * VX + x) * 32;
    float d = sigm(bf2f(row[17]));
    float w = d * trans;
    trans *= (1.f - d);
#pragma unroll
    for (int c = 0; c < 17; ++c) seg[c] += w * bf2f(row[c]);
#pragma unroll
    for (int j = 0; j < 3; ++j) rgb[j] += w * sigm(bf2f(row[18 + j]));
  }
  size_t ob = (size_t)b * 100 * VY * VX + (size_t)y * VX + x;
#pragma unroll
  for (int c = 0; c < 17; ++c) out[ob + (size_t)(80 + c) * VY * VX] = seg[c];
#pragma unroll
  for (int j = 0; j < 3; ++j) out[ob + (size_t)(97 + j) * VY * VX] = rgb[j];
}

// ---------------- 1x1 conv (K=1280) via WMMA ----------------
__global__ __launch_bounds__(256) void k_bev(const bf16* feat, const bf16* wpk,
                                             const float* b_out, float* out) {
  int wv = (blockIdx.x * blockDim.x + threadIdx.x) >> 5;
  int lane = threadIdx.x & 31;
  const int NW = BATCH * VY * (VX / 16); // 2048 waves
  if (wv >= NW) return;
  int xt = wv & 7; int y = (wv >> 3) & 127; int b = wv >> 10;
  int col = lane & 15, kb = (lane >> 4) * 8, rowb = kb;
  v8f acc[5];
#pragma unroll
  for (int c = 0; c < 5; ++c)
#pragma unroll
    for (int i = 0; i < 8; ++i) acc[c][i] = 0.f;

  const bf16* prow = feat + ((size_t)(b * VY + y) * VX + xt * 16 + col) * (MID * VZ);
  for (int kc = 0; kc < 40; ++kc) {
    v16bf bf = load_frag_row(prow + kc * 32, kb);
    __builtin_prefetch(prow + kc * 32 + 128, 0, 0);
#pragma unroll
    for (int c = 0; c < 5; ++c) {
      const v16bf* ap =
          (const v16bf*)(wpk + (((size_t)c * 40 + kc) * 32 + lane) * 16);
      acc[c] = wmma_bf16(*ap, bf, acc[c]);
    }
  }
  int xx = xt * 16 + col;
#pragma unroll
  for (int c = 0; c < 5; ++c)
#pragma unroll
    for (int v = 0; v < 8; ++v) {
      int co = c * 16 + rowb + v;
      out[((size_t)(b * 100 + co) * VY + y) * VX + xx] = acc[c][v] + b_out[co];
    }
}

// ---------------- host ----------------
extern "C" void kernel_launch(void* const* d_in, const int* in_sizes, int n_in,
                              void* d_out, int out_size, void* d_ws, size_t ws_size,
                              hipStream_t stream) {
  (void)in_sizes; (void)n_in; (void)out_size; (void)ws_size;
  const float* img       = (const float*)d_in[0];
  const float* s2e       = (const float*)d_in[1];
  const float* intr      = (const float*)d_in[2];
  const float* ida       = (const float*)d_in[3];
  const float* bda       = (const float*)d_in[4];
  const float* w_lower   = (const float*)d_in[5];
  const float* w_base    = (const float*)d_in[6];
  const float* b_base    = (const float*)d_in[7];
  const float* w_density = (const float*)d_in[8];
  const float* b_density = (const float*)d_in[9];
  const float* w_seg     = (const float*)d_in[10];
  const float* b_seg     = (const float*)d_in[11];
  const float* w_feat    = (const float*)d_in[12];
  const float* b_feat    = (const float*)d_in[13];
  const float* w_rgb     = (const float*)d_in[14];
  const float* b_rgb     = (const float*)d_in[15];
  const float* w_out     = (const float*)d_in[16];
  const float* b_out     = (const float*)d_in[17];
  float* out = (float*)d_out;
  char* ws = (char*)d_ws;

  float* pmat  = (float*)(ws + OFF_PMAT);
  bf16* wlow   = (bf16*)(ws + OFF_WLOW);
  bf16* wbase  = (bf16*)(ws + OFF_WBASE);
  bf16* wfeat  = (bf16*)(ws + OFF_WFEAT);
  bf16* wsrd   = (bf16*)(ws + OFF_WSRD);
  bf16* wout   = (bf16*)(ws + OFF_WOUT);
  bf16* imgcl  = (bf16*)(ws + OFF_IMGCL);
  float* xcl   = (float*)(ws + OFF_XCL);
  bf16* vox    = (bf16*)(ws + OFF_VOX);
  bf16* basecl = (bf16*)(ws + OFF_BASECL);
  bf16* srd    = (bf16*)(ws + OFF_SRD);
  bf16* featcl = (bf16*)(ws + OFF_FEAT);

  // preprocessing
  k_build_proj<<<1, 16, 0, stream>>>(s2e, intr, bda, pmat);
  {
    int total = BATCH * NCAM * FH * FW * CIN;
    k_img_cl<<<(total + 255) / 256, 256, 0, stream>>>(img, imgcl);
  }
  {
    int total = BATCH * NCAM * FH * 6 * CIN;
    k_img_halo<<<(total + 255) / 256, 256, 0, stream>>>(imgcl);
  }
  {
    int total = BATCH * VZ * VY * 2 * CPAD;
    k_halo_zero<<<(total + 255) / 256, 256, 0, stream>>>(vox, BATCH * VZ * VY, VXH,
                                                         CPAD);
    k_halo_zero<<<(total + 255) / 256, 256, 0, stream>>>(basecl, BATCH * VZ * VY,
                                                         VXH, CPAD);
  }
  k_pack_lower<<<(5 * 9 * 8 * 512 + 255) / 256, 256, 0, stream>>>(w_lower, wlow);
  k_pack_c3d<<<(5 * 27 * 3 * 512 + 255) / 256, 256, 0, stream>>>(w_base, wbase);
  k_pack_c3d<<<(5 * 27 * 3 * 512 + 255) / 256, 256, 0, stream>>>(w_feat, wfeat);
  k_pack_srd<<<(2 * 27 * 3 * 512 + 255) / 256, 256, 0, stream>>>(w_seg, w_density,
                                                                 w_rgb, wsrd);
  k_pack_out<<<(5 * 40 * 512 + 255) / 256, 256, 0, stream>>>(w_out, wout);

  // channel_lower conv2d (WMMA)
  k_conv_lower<<<(BATCH * NCAM * 5 * FH * 3) / 8, 256, 0, stream>>>(imgcl, wlow, xcl);

  // projection + bilinear gather -> vox
  {
    int total = BATCH * VZ * VY * VX * 6;
    k_gather<<<(total + 255) / 256, 256, 0, stream>>>(xcl, pmat, ida, vox);
  }

  const int c3d_blocks = (BATCH * VZ * VY * (VX / 16)) / 8; // 4096
  k_conv3d<5, 0><<<c3d_blocks, 256, 0, stream>>>(vox, wbase, b_base, nullptr,
                                                 nullptr, basecl);
  k_conv3d<2, 1><<<c3d_blocks, 256, 0, stream>>>(basecl, wsrd, b_seg, b_density,
                                                 b_rgb, srd);
  k_conv3d<5, 2><<<c3d_blocks, 256, 0, stream>>>(basecl, wfeat, b_feat, nullptr,
                                                 nullptr, featcl);

  // compositing -> out channels 80..99
  k_composite<<<(BATCH * VY * VX) / 256, 256, 0, stream>>>(srd, out);

  // 1x1 conv (K=1280) -> out channels 0..79
  k_bev<<<(BATCH * VY * (VX / 16)) / 8, 256, 0, stream>>>(featcl, wout, b_out, out);
}
